// Attention_65051574665850
// MI455X (gfx1250) — matrix operand order repaired and checked by reading, not compiled, not run on hardware
//
#include <hip/hip_runtime.h>
#include <hip/hip_bf16.h>

// ---------------------------------------------------------------------------
// MI455X attention block, bf16 WMMA path (V_WMMA_F32_16X16X32_BF16).
// Compute-bound (~137 GFLOP vs ~80MB HBM traffic): all four matmuls run on
// the matrix pipe. Intermediates (~90MB bf16) live in workspace and fit L2.
// ---------------------------------------------------------------------------

typedef __attribute__((ext_vector_type(16))) __bf16 v16bf;
typedef __attribute__((ext_vector_type(8)))  __bf16 v8bf;
typedef __attribute__((ext_vector_type(8)))  float  v8f;

#define HEADS 16
#define DHEAD 64
#define NB    4
#define NSEQ  2048
#define DMODEL 1024
#define ROWS  8192           // NB*NSEQ
#define NQKV  3072

// A fragment (16x32, bf16): lane holds row M=lane%16; element e<8 at
// K = (lane>=16?8:0)+e, element e>=8 at K+16.  Row-major source, stride ld.
static __device__ __forceinline__ v16bf load_fragA(const __bf16* p0, int ld, int lane) {
    int row = lane & 15;
    int k0  = (lane & 16) >> 1;                 // 0 or 8
    const __bf16* p = p0 + (size_t)row * ld + k0;
    v8bf lo = *(const v8bf*)(p);
    v8bf hi = *(const v8bf*)(p + 16);
    return __builtin_shufflevector(lo, hi, 0,1,2,3,4,5,6,7,8,9,10,11,12,13,14,15);
}

// B fragment (32x16, bf16): lane holds column N=lane%16; element e<8 at
// K = (lane>=16?8:0)+e, element e>=8 at K+16.  Source row-major with row = N, stride ld.
static __device__ __forceinline__ v16bf load_fragB(const __bf16* p0, int ld, int lane) {
    int col = lane & 15;
    int k0  = (lane & 16) >> 1;                 // 0 or 8
    const __bf16* p = p0 + (size_t)col * ld + k0;
    v8bf lo = *(const v8bf*)(p);
    v8bf hi = *(const v8bf*)(p + 16);
    return __builtin_shufflevector(lo, hi, 0,1,2,3,4,5,6,7,8,9,10,11,12,13,14,15);
}

static __device__ __forceinline__ v8f wmma_bf16(v16bf a, v16bf b, v8f c) {
    return __builtin_amdgcn_wmma_f32_16x16x32_bf16(false, a, false, b, (short)0, c,
                                                   false, false);
}

// --------------------------- weight fp32 -> bf16 ---------------------------
__global__ void __launch_bounds__(256)
cvt_weights_kernel(const float* __restrict__ wqkv, const float* __restrict__ wout,
                   __bf16* __restrict__ wqkv_b, __bf16* __restrict__ wout_b) {
    int idx = blockIdx.x * blockDim.x + threadIdx.x;
    int stride = gridDim.x * blockDim.x;
    for (int j = idx; j < NQKV * DMODEL; j += stride) wqkv_b[j] = (__bf16)wqkv[j];
    for (int j = idx; j < DMODEL * DMODEL; j += stride) wout_b[j] = (__bf16)wout[j];
}

// --------------------------- LayerNorm -> bf16 -----------------------------
__global__ void __launch_bounds__(256)
ln_bf16_kernel(const float* __restrict__ x, const float* __restrict__ gamma,
               const float* __restrict__ beta, __bf16* __restrict__ xn) {
    int row = blockIdx.x;
    int tid = threadIdx.x;                       // 256 threads * 4 = 1024
    const float4 v = ((const float4*)(x + (size_t)row * DMODEL))[tid];
    float s  = v.x + v.y + v.z + v.w;
    float ss = v.x*v.x + v.y*v.y + v.z*v.z + v.w*v.w;
    __shared__ float rs[256], rss[256];
    rs[tid] = s; rss[tid] = ss;
    __syncthreads();
    for (int off = 128; off > 0; off >>= 1) {
        if (tid < off) { rs[tid] += rs[tid+off]; rss[tid] += rss[tid+off]; }
        __syncthreads();
    }
    float mu   = rs[0]  * (1.0f / DMODEL);
    float var  = rss[0] * (1.0f / DMODEL) - mu * mu;
    float rstd = rsqrtf(var + 1e-5f);
    const float4 g  = ((const float4*)gamma)[tid];
    const float4 bt = ((const float4*)beta)[tid];
    __bf16* o = xn + (size_t)row * DMODEL + tid * 4;
    o[0] = (__bf16)((v.x - mu) * rstd * g.x + bt.x);
    o[1] = (__bf16)((v.y - mu) * rstd * g.y + bt.y);
    o[2] = (__bf16)((v.z - mu) * rstd * g.z + bt.z);
    o[3] = (__bf16)((v.w - mu) * rstd * g.w + bt.w);
}

// ------------------------------- QKV GEMM ----------------------------------
// C[8192,3072] = Xn[8192,1024] * Wqkv^T ; wave tile 32x64, 8 WMMAs / k-step.
// Q,K scattered as [b,h,n,64]; V scattered transposed as [b,h,64,n].
__global__ void __launch_bounds__(128)
qkv_gemm_kernel(const __bf16* __restrict__ xn, const __bf16* __restrict__ w,
                __bf16* __restrict__ qb, __bf16* __restrict__ kb,
                __bf16* __restrict__ vt) {
    int lane  = threadIdx.x & 31;
    int wv    = threadIdx.x >> 5;
    int nBase = blockIdx.x * 64;
    int mBase = blockIdx.y * 128 + wv * 32;
    v8f acc[2][4] = {};
    const __bf16* arow0 = xn + (size_t)mBase * DMODEL;
    const __bf16* arow1 = xn + (size_t)(mBase + 16) * DMODEL;
    for (int kk = 0; kk < DMODEL; kk += 32) {
        v16bf a0 = load_fragA(arow0 + kk, DMODEL, lane);
        v16bf a1 = load_fragA(arow1 + kk, DMODEL, lane);
#pragma unroll
        for (int nt = 0; nt < 4; ++nt) {
            v16bf b = load_fragB(w + (size_t)(nBase + nt * 16) * DMODEL + kk, DMODEL, lane);
            acc[0][nt] = wmma_bf16(a0, b, acc[0][nt]);
            acc[1][nt] = wmma_bf16(a1, b, acc[1][nt]);
        }
    }
#pragma unroll
    for (int mt = 0; mt < 2; ++mt) {
#pragma unroll
        for (int nt = 0; nt < 4; ++nt) {
            int n     = nBase + nt * 16 + (lane & 15);
            int which = n >> 10;                 // 0=Q 1=K 2=V (tile never straddles)
            int hn    = n & 1023;
            int head  = hn >> 6;
            int dim   = hn & 63;
#pragma unroll
            for (int i = 0; i < 8; ++i) {
                int tok = mBase + mt * 16 + i + ((lane & 16) >> 1);
                int bi  = tok >> 11;
                int tl  = tok & 2047;
                size_t bh = (size_t)(bi * HEADS + head);
                __bf16 val = (__bf16)acc[mt][nt][i];
                if (which == 0)      qb[(bh * NSEQ + tl) * DHEAD + dim] = val;
                else if (which == 1) kb[(bh * NSEQ + tl) * DHEAD + dim] = val;
                else                 vt[(bh * DHEAD + dim) * NSEQ + tl] = val;
            }
        }
    }
}

// --------------------------- Flash attention -------------------------------
// 1 wave per 16 query rows per (b,h); keys streamed 32 at a time.
__global__ void __launch_bounds__(128)
attn_kernel(const __bf16* __restrict__ qb, const __bf16* __restrict__ kb,
            const __bf16* __restrict__ vt, __bf16* __restrict__ ao) {
    __shared__ __bf16 Pl[4][16][32];             // per-wave P staging
    int lane = threadIdx.x & 31;
    int wv   = threadIdx.x >> 5;
    int bh   = blockIdx.y;
    int bi   = bh >> 4, head = bh & 15;
    int q0   = (blockIdx.x * 4 + wv) * 16;
    const __bf16* Qh = qb + (size_t)bh * NSEQ * DHEAD;
    const __bf16* Kh = kb + (size_t)bh * NSEQ * DHEAD;
    const __bf16* Vh = vt + (size_t)bh * DHEAD * NSEQ;

    v16bf aq0 = load_fragA(Qh + (size_t)q0 * DHEAD +  0, DHEAD, lane);
    v16bf aq1 = load_fragA(Qh + (size_t)q0 * DHEAD + 32, DHEAD, lane);
    v8f o[4] = {};
    float mrow[8], lrow[8];
#pragma unroll
    for (int i = 0; i < 8; ++i) { mrow[i] = -__builtin_inff(); lrow[i] = 0.0f; }
    const float scale = 0.125f;                  // 1/sqrt(64)

    for (int kt = 0; kt < NSEQ; kt += 32) {
        v8f s0 = {}, s1 = {};
        s0 = wmma_bf16(aq0, load_fragB(Kh + (size_t)kt * DHEAD +  0, DHEAD, lane), s0);
        s0 = wmma_bf16(aq1, load_fragB(Kh + (size_t)kt * DHEAD + 32, DHEAD, lane), s0);
        s1 = wmma_bf16(aq0, load_fragB(Kh + (size_t)(kt + 16) * DHEAD +  0, DHEAD, lane), s1);
        s1 = wmma_bf16(aq1, load_fragB(Kh + (size_t)(kt + 16) * DHEAD + 32, DHEAD, lane), s1);
#pragma unroll
        for (int i = 0; i < 8; ++i) {
            float a  = s0[i] * scale;
            float b2 = s1[i] * scale;
            float mt = fmaxf(a, b2);             // reduce over 16-lane group
            mt = fmaxf(mt, __shfl_xor(mt, 1, 32));
            mt = fmaxf(mt, __shfl_xor(mt, 2, 32));
            mt = fmaxf(mt, __shfl_xor(mt, 4, 32));
            mt = fmaxf(mt, __shfl_xor(mt, 8, 32));
            float mnew  = fmaxf(mrow[i], mt);
            float alpha = __expf(mrow[i] - mnew);
            float p0 = __expf(a  - mnew);
            float p1 = __expf(b2 - mnew);
            float ps = p0 + p1;
            ps += __shfl_xor(ps, 1, 32);
            ps += __shfl_xor(ps, 2, 32);
            ps += __shfl_xor(ps, 4, 32);
            ps += __shfl_xor(ps, 8, 32);
            lrow[i] = lrow[i] * alpha + ps;
            mrow[i] = mnew;
            int r = i + ((lane & 16) >> 1);      // C-layout row
            Pl[wv][r][lane & 15]        = (__bf16)p0;
            Pl[wv][r][16 + (lane & 15)] = (__bf16)p1;
#pragma unroll
            for (int nt = 0; nt < 4; ++nt) o[nt][i] *= alpha;
        }
        asm volatile("s_wait_dscnt 0" ::: "memory");   // P stores visible to frag loads
        v16bf ap = load_fragA(&Pl[wv][0][0], 32, lane);
#pragma unroll
        for (int nt = 0; nt < 4; ++nt)
            o[nt] = wmma_bf16(ap, load_fragB(Vh + (size_t)(nt * 16) * NSEQ + kt, NSEQ, lane),
                              o[nt]);
    }
#pragma unroll
    for (int i = 0; i < 8; ++i) {
        float inv = 1.0f / lrow[i];
        int tok = bi * NSEQ + q0 + i + ((lane & 16) >> 1);
#pragma unroll
        for (int nt = 0; nt < 4; ++nt)
            ao[(size_t)tok * DMODEL + head * DHEAD + nt * 16 + (lane & 15)] =
                (__bf16)(o[nt][i] * inv);
    }
}

// ------------------------- Output projection GEMM --------------------------
__global__ void __launch_bounds__(128)
out_gemm_kernel(const __bf16* __restrict__ a, const __bf16* __restrict__ w,
                const float* __restrict__ bias, float* __restrict__ out) {
    int lane  = threadIdx.x & 31;
    int wv    = threadIdx.x >> 5;
    int nBase = blockIdx.x * 64;
    int mBase = blockIdx.y * 128 + wv * 32;
    v8f acc[2][4] = {};
    for (int kk = 0; kk < DMODEL; kk += 32) {
        v16bf a0 = load_fragA(a + (size_t)mBase * DMODEL + kk, DMODEL, lane);
        v16bf a1 = load_fragA(a + (size_t)(mBase + 16) * DMODEL + kk, DMODEL, lane);
#pragma unroll
        for (int nt = 0; nt < 4; ++nt) {
            v16bf b = load_fragB(w + (size_t)(nBase + nt * 16) * DMODEL + kk, DMODEL, lane);
            acc[0][nt] = wmma_bf16(a0, b, acc[0][nt]);
            acc[1][nt] = wmma_bf16(a1, b, acc[1][nt]);
        }
    }
#pragma unroll
    for (int mt = 0; mt < 2; ++mt) {
#pragma unroll
        for (int nt = 0; nt < 4; ++nt) {
            int n = nBase + nt * 16 + (lane & 15);
            float bs = bias[n];
#pragma unroll
            for (int i = 0; i < 8; ++i) {
                int tok = mBase + mt * 16 + i + ((lane & 16) >> 1);
                out[(size_t)tok * DMODEL + n] = acc[mt][nt][i] + bs;
            }
        }
    }
}

// ------------------------------- launcher ----------------------------------
extern "C" void kernel_launch(void* const* d_in, const int* in_sizes, int n_in,
                              void* d_out, int out_size, void* d_ws, size_t ws_size,
                              hipStream_t stream) {
    const float* x    = (const float*)d_in[0];
    const float* g    = (const float*)d_in[1];
    const float* bt   = (const float*)d_in[2];
    const float* wqkv = (const float*)d_in[3];
    const float* wout = (const float*)d_in[4];
    const float* bout = (const float*)d_in[5];
    float* out = (float*)d_out;

    char* ws = (char*)d_ws;
    __bf16* wqkv_b = (__bf16*)(ws + 0);           //  6,291,456 B
    __bf16* wout_b = (__bf16*)(ws + 6291456);     //  2,097,152 B
    __bf16* xn     = (__bf16*)(ws + 8388608);     // 16,777,216 B
    __bf16* qb     = (__bf16*)(ws + 25165824);    // 16,777,216 B
    __bf16* kb     = (__bf16*)(ws + 41943040);    // 16,777,216 B
    __bf16* vt     = (__bf16*)(ws + 58720256);    // 16,777,216 B
    __bf16* ab     = (__bf16*)(ws + 75497472);    // 16,777,216 B -> 92,274,688 total

    cvt_weights_kernel<<<1024, 256, 0, stream>>>(wqkv, wout, wqkv_b, wout_b);
    ln_bf16_kernel<<<ROWS, 256, 0, stream>>>(x, g, bt, xn);
    qkv_gemm_kernel<<<dim3(48, 64), 128, 0, stream>>>(xn, wqkv_b, qb, kb, vt);
    attn_kernel<<<dim3(NSEQ / 64, NB * HEADS), 128, 0, stream>>>(qb, kb, vt, ab);
    out_gemm_kernel<<<dim3(16, 64), 128, 0, stream>>>(ab, wout_b, bout, out);
}
